// SimbaBlock_68788196213187
// MI455X (gfx1250) — matrix operand order repaired and checked by reading, not compiled, not run on hardware
//
#include <hip/hip_runtime.h>
#include <hip/hip_bf16.h>

// ---------------------------------------------------------------------------
// SiMBA block for MI455X (gfx1250, wave32, WMMA).
// All dense contractions -> v_wmma_f32_16x16x32_f16 (f16 in, f32 accumulate),
// register-blocked 2x4 tiles per wave (8 independent WMMAs per K-step).
// Loads indexed off the kernel-arg base pointer (no pointer phis) so they
// lower to global_load_b128 with immediate offsets, not flat_load.
// ---------------------------------------------------------------------------

#define B_    2
#define S_    512
#define DIM_  512
#define DIN_  1024
#define DSTATE_ 64
#define DCONV_  64
#define DTRANK_ 32
#define NB_   4
#define BS_   128
#define LAM_  0.01f
#define M_    (B_ * S_)      // 1024 rows for all token-parallel GEMMs

typedef _Float16 v8h  __attribute__((ext_vector_type(8)));
typedef _Float16 v16h __attribute__((ext_vector_type(16)));
typedef float    v8f  __attribute__((ext_vector_type(8)));

// ---------------------------------------------------------------------------
// Generic WMMA GEMM:  C[M,N](f32) = act( A[M,K](f16) @ B^T  + bias ) (+resid)
// B stored transposed (N x K, f16) so both fragments are contiguous 16B loads
// using a contiguous k order (same for A and B) in V_WMMA_F32_16X16X32_F16:
//   A lane l: row m = l&15, K = (l>>4)*16 + e                   (one 16-half run)
//   B lane l: col n = l&15, K = (l>>4)*16 + e                   (one 16-half run)
//   C lane l: elem r -> row = 16*tm + r + (l>=16 ? 8 : 0), col = 16*tn + (l&15)
// Each wave owns an MW x NW block of 16x16 tiles: MW A-frags + NW B-frags feed
// MW*NW independent WMMAs per 32-wide K-step (fragment reuse, latency hiding).
// grid.z = independent blocks (EinFFT), flat super-tile id = blockIdx.x*8+wave.
// act: 0=none 1=softplus 2=relu 3=softshrink
// ---------------------------------------------------------------------------
template <int MW, int NW>
__global__ __launch_bounds__(256) void gemm_wmma_f16(
    const _Float16* __restrict__ A, int lda, long long sAz,
    const _Float16* __restrict__ Bt, int ldb, long long sBz,
    float* __restrict__ C, int ldc, long long sCz,
    const float* __restrict__ bias, long long sbz,
    const float* __restrict__ resid,
    _Float16* __restrict__ Ch,
    int M, int N, int K, int act)
{
    const int lane = threadIdx.x & 31;
    const int wave = threadIdx.x >> 5;
    const int snx  = N / (16 * NW);          // super-tiles along N
    const int wt   = blockIdx.x * 8 + wave;  // flat super-tile id
    const int tmg  = wt / snx;
    const int tng  = wt - tmg * snx;
    if (tmg >= M / (16 * MW)) return;

    const _Float16* Az = A  + (size_t)blockIdx.z * sAz;
    const _Float16* Bz = Bt + (size_t)blockIdx.z * sBz;
    float*          Cz = C  + (size_t)blockIdx.z * sCz;
    _Float16*       Hz = Ch ? Ch + (size_t)blockIdx.z * sCz : ((_Float16*)0);

    const int l15 = lane & 15;
    const int akb = (lane >> 4) * 16;   // A k-sub-base within 32-wide step
    const int bkb = (lane >> 4) * 16;   // B k-sub-base

    // Fixed per-lane element offsets; loads index base + off + k (no ptr phis).
    size_t aoff[MW], boff[NW];
    #pragma unroll
    for (int i = 0; i < MW; ++i)
        aoff[i] = (size_t)((tmg * MW + i) * 16 + l15) * lda + akb;
    #pragma unroll
    for (int j = 0; j < NW; ++j)
        boff[j] = (size_t)((tng * NW + j) * 16 + l15) * ldb + bkb;

    v8f acc[MW][NW];
    #pragma unroll
    for (int i = 0; i < MW; ++i)
        #pragma unroll
        for (int j = 0; j < NW; ++j) acc[i][j] = (v8f){};

    for (int k = 0; k < K; k += 32) {
        v16h av[MW], bv[NW];
        #pragma unroll
        for (int i = 0; i < MW; ++i) {
            const _Float16* p = Az + aoff[i] + k;
            __builtin_prefetch(p + 64, 0, 1);       // -> global_prefetch_b8
            v8h a0 = *(const v8h*)(p);              // K = akb+0..7
            v8h a1 = *(const v8h*)(p + 8);          // K = akb+8..15
            av[i] = __builtin_shufflevector(a0, a1, 0,1,2,3,4,5,6,7,8,9,10,11,12,13,14,15);
        }
        #pragma unroll
        for (int j = 0; j < NW; ++j) {
            const _Float16* p = Bz + boff[j] + k;
            __builtin_prefetch(p + 64, 0, 1);
            v8h b0 = *(const v8h*)(p);              // K = bkb+0..7
            v8h b1 = *(const v8h*)(p + 8);          // K = bkb+8..15
            bv[j] = __builtin_shufflevector(b0, b1, 0,1,2,3,4,5,6,7,8,9,10,11,12,13,14,15);
        }
        #pragma unroll
        for (int i = 0; i < MW; ++i)
            #pragma unroll
            for (int j = 0; j < NW; ++j)
                acc[i][j] = __builtin_amdgcn_wmma_f32_16x16x32_f16(
                                false, av[i], false, bv[j], (short)0, acc[i][j],
                                false, false);
    }

    #pragma unroll
    for (int j = 0; j < NW; ++j) {
        const int col = (tng * NW + j) * 16 + l15;
        const float bvs = bias ? bias[(size_t)blockIdx.z * sbz + col] : 0.f;
        #pragma unroll
        for (int i = 0; i < MW; ++i) {
            #pragma unroll
            for (int r = 0; r < 8; ++r) {
                int row = (tmg * MW + i) * 16 + r + ((lane >> 4) * 8);
                float v = acc[i][j][r] + bvs;
                if (act == 1) v = (v > 20.f) ? v : log1pf(__expf(v));        // softplus
                else if (act == 2) v = fmaxf(v, 0.f);                        // relu
                else if (act == 3) v = (v >  LAM_) ? v - LAM_ :
                                       (v < -LAM_) ? v + LAM_ : 0.f;         // softshrink
                size_t o = (size_t)row * ldc + col;
                if (resid) v += resid[(size_t)blockIdx.z * sCz + o];
                Cz[o] = v;
                if (Hz) Hz[o] = (_Float16)v;
            }
        }
    }
}

// ---------------------------------------------------------------------------
// Weight prep: f32 (K,N) row-major  ->  f16 (N,K)  (transposed cast)
// ---------------------------------------------------------------------------
__global__ void transpose_cast_f16(const float* __restrict__ src,
                                   _Float16* __restrict__ dst, int K, int N)
{
    int idx = blockIdx.x * 256 + threadIdx.x;
    if (idx >= K * N) return;
    int k = idx / N, n = idx - k * N;
    dst[(size_t)n * K + k] = (_Float16)src[idx];
}

// EinFFT stacked complex weights: per block, Wt (N=256, K=256) f16 with
//   out_r (n<128):  k<128 -> Wr[k][n],           k>=128 -> -Wi[k-128][n]
//   out_i (n>=128): k<128 -> Wi[k][n-128],       k>=128 ->  Wr[k-128][n-128]
// bias stack (NB,256) = [br | bi]
__global__ void build_einfft_wt(const float* __restrict__ cw,   // (2,NB,128,128)
                                const float* __restrict__ cb,   // (2,NB,128)
                                _Float16* __restrict__ Wt,      // (NB,256,256)
                                float* __restrict__ bstack)     // (NB,256)
{
    int idx = blockIdx.x * 256 + threadIdx.x;
    if (idx >= NB_ * 256 * 256) return;
    int nb  = idx / (256 * 256);
    int rem = idx - nb * 256 * 256;
    int n = rem >> 8, k = rem & 255;
    const float* Wr = cw + (size_t)(0 * NB_ + nb) * 128 * 128;
    const float* Wi = cw + (size_t)(1 * NB_ + nb) * 128 * 128;
    float v;
    if (n < 128) v = (k < 128) ?  Wr[k * 128 + n]          : -Wi[(k - 128) * 128 + n];
    else         v = (k < 128) ?  Wi[k * 128 + (n - 128)]  :  Wr[(k - 128) * 128 + (n - 128)];
    Wt[idx] = (_Float16)v;
    if (idx < NB_ * 256) {
        int nb2 = idx >> 8, n2 = idx & 255;
        bstack[idx] = (n2 < 128) ? cb[(0 * NB_ + nb2) * 128 + n2]
                                 : cb[(1 * NB_ + nb2) * 128 + (n2 - 128)];
    }
}

// Slice columns [soff, soff+cols) of f32 (rows, sld) -> f16 (rows, dld)
__global__ void slice_cast_f16(const float* __restrict__ src, int sld, int soff,
                               _Float16* __restrict__ dst, int dld,
                               int rows, int cols)
{
    int idx = blockIdx.x * 256 + threadIdx.x;
    if (idx >= rows * cols) return;
    int r = idx / cols, c = idx - r * cols;
    dst[(size_t)r * dld + c] = (_Float16)src[(size_t)r * sld + soff + c];
}

// ---------------------------------------------------------------------------
// LayerNorm over last dim (=512), one block per row, 256 threads.
// ---------------------------------------------------------------------------
__global__ __launch_bounds__(256) void layernorm_k(const float* __restrict__ x,
                                                   float* __restrict__ outf,
                                                   _Float16* __restrict__ outh)
{
    __shared__ float s1[256], s2[256];
    const int row = blockIdx.x, tid = threadIdx.x;
    const float* xr = x + (size_t)row * DIM_;
    float a = xr[tid], b = xr[tid + 256];
    s1[tid] = a + b;
    s2[tid] = a * a + b * b;
    __syncthreads();
    for (int off = 128; off > 0; off >>= 1) {
        if (tid < off) { s1[tid] += s1[tid + off]; s2[tid] += s2[tid + off]; }
        __syncthreads();
    }
    float mean = s1[0] * (1.f / DIM_);
    float var  = s2[0] * (1.f / DIM_) - mean * mean;
    float rs   = rsqrtf(var + 1e-5f);
    float v0 = (a - mean) * rs, v1 = (b - mean) * rs;
    if (outf) { outf[(size_t)row * DIM_ + tid] = v0; outf[(size_t)row * DIM_ + tid + 256] = v1; }
    if (outh) { outh[(size_t)row * DIM_ + tid] = (_Float16)v0; outh[(size_t)row * DIM_ + tid + 256] = (_Float16)v1; }
}

// ---------------------------------------------------------------------------
// Causal depthwise conv1d (kernel 64, left pad 63) + bias + SiLU.
// One block per (b, d) column of xm (= first DIN_ cols of in_proj output P).
// ---------------------------------------------------------------------------
__global__ __launch_bounds__(128) void conv_silu_k(const float* __restrict__ P,
                                                   const float* __restrict__ cw,
                                                   const float* __restrict__ cb,
                                                   float* __restrict__ u,
                                                   _Float16* __restrict__ uh)
{
    const int bd = blockIdx.x;
    const int b = bd >> 10, d = bd & 1023;
    __shared__ float col[DCONV_ - 1 + S_];
    __shared__ float w[DCONV_];
    const int tid = threadIdx.x;
    for (int i = tid; i < DCONV_ - 1; i += 128) col[i] = 0.f;
    for (int s = tid; s < S_; s += 128)
        col[DCONV_ - 1 + s] = P[((size_t)(b * S_ + s)) * (2 * DIN_) + d];
    if (tid < DCONV_) w[tid] = cw[d * DCONV_ + tid];
    __syncthreads();
    const float bias = cb[d];
    for (int s = tid; s < S_; s += 128) {
        float acc = bias;
        #pragma unroll
        for (int k = 0; k < DCONV_; ++k) acc += w[k] * col[s + k];
        float uv = acc / (1.f + __expf(-acc));     // silu
        size_t o = ((size_t)(b * S_ + s)) * DIN_ + d;
        u[o] = uv;
        uh[o] = (_Float16)uv;
    }
}

// ---------------------------------------------------------------------------
// Selective-scan. One lane per (b, d): 64 states held in registers (wave32,
// fully unrolled so h[]/A[] stay in VGPRs). Fuses  (y + u*D) * silu(res).
// ---------------------------------------------------------------------------
__global__ __launch_bounds__(256) void ssm_scan_k(const float* __restrict__ dt,
                                                  const float* __restrict__ uu,
                                                  const float* __restrict__ xdbl,
                                                  const float* __restrict__ A_log,
                                                  const float* __restrict__ Dw,
                                                  const float* __restrict__ P,
                                                  _Float16* __restrict__ yh)
{
    const int g = blockIdx.x * 256 + threadIdx.x;   // 0..2047
    const int b = g >> 10, d = g & 1023;
    float A[DSTATE_], h[DSTATE_];
    #pragma unroll
    for (int n = 0; n < DSTATE_; ++n) {
        A[n] = -__expf(A_log[d * DSTATE_ + n]);
        h[n] = 0.f;
    }
    const float Dd = Dw[d];
    for (int s = 0; s < S_; ++s) {
        const size_t r = (size_t)(b * S_ + s);
        const float delta = dt[r * DIN_ + d];
        const float uv    = uu[r * DIN_ + d];
        const float du    = delta * uv;
        const float* Brow = xdbl + r * (DTRANK_ + 2 * DSTATE_) + DTRANK_;
        const float* Crow = Brow + DSTATE_;
        float y = 0.f;
        #pragma unroll
        for (int n = 0; n < DSTATE_; ++n) {
            float hn = __expf(delta * A[n]) * h[n] + du * Brow[n];
            h[n] = hn;
            y += hn * Crow[n];
        }
        const float rv = P[r * (2 * DIN_) + DIN_ + d];
        const float gate = rv / (1.f + __expf(-rv));
        yh[r * DIN_ + d] = (_Float16)((y + uv * Dd) * gate);
    }
}

// ---------------------------------------------------------------------------
// 512-point radix-2 FFT in LDS. 256 threads = 256 butterflies / stage.
// pos = 2*tid - (tid & (half-1)); disjoint butterflies -> barrier per stage.
// ---------------------------------------------------------------------------
__device__ inline void fft512_stages(float* re, float* im, int tid, float sign)
{
    for (int len = 2; len <= 512; len <<= 1) {
        const int half = len >> 1;
        const int k = tid & (half - 1);
        const int pos = (tid << 1) - k;
        float s, c;
        __sincosf(sign * 6.283185307179586f * (float)k / (float)len, &s, &c);
        float xr = re[pos + half], xi = im[pos + half];
        float tr = c * xr - s * xi;
        float ti = c * xi + s * xr;
        float ur = re[pos], ui = im[pos];
        re[pos] = ur + tr;         im[pos] = ui + ti;
        re[pos + half] = ur - tr;  im[pos + half] = ui - ti;
        __syncthreads();
    }
}

// Forward FFT (ortho) of one (b,d) column of ln2; writes f16 [Xr|Xi] per block
// into Xcat (row = b*S+f, layout (M, NB, 256)) feeding the EinFFT GEMM.
__global__ __launch_bounds__(256) void fft_fwd_k(const float* __restrict__ xin,
                                                 _Float16* __restrict__ Xcat)
{
    const int bd = blockIdx.x;
    const int b = bd >> 9, d = bd & 511;
    __shared__ float re[512], im[512];
    const int tid = threadIdx.x;
    for (int i = tid; i < 512; i += 256) {
        int r = (int)(__brev((unsigned)i) >> 23);   // 9-bit reversal
        re[r] = xin[((size_t)(b * S_ + i)) * DIM_ + d];
        im[r] = 0.f;
    }
    __syncthreads();
    fft512_stages(re, im, tid, -1.f);
    const float sc = 0.044194173824159216f;         // 1/sqrt(512), ortho
    const int nb = d >> 7, j = d & 127;
    for (int f = tid; f < 512; f += 256) {
        size_t row = (size_t)(b * S_ + f);
        Xcat[row * (NB_ * 256) + nb * 256 + j]       = (_Float16)(re[f] * sc);
        Xcat[row * (NB_ * 256) + nb * 256 + 128 + j] = (_Float16)(im[f] * sc);
    }
}

// Inverse FFT (ortho) of Zcat [Zr|Zi], take real, add residual, write d_out.
__global__ __launch_bounds__(256) void ifft_out_k(const float* __restrict__ Zcat,
                                                  const float* __restrict__ outres,
                                                  float* __restrict__ dout)
{
    const int bd = blockIdx.x;
    const int b = bd >> 9, d = bd & 511;
    __shared__ float re[512], im[512];
    const int tid = threadIdx.x;
    const int nb = d >> 7, j = d & 127;
    for (int i = tid; i < 512; i += 256) {
        int r = (int)(__brev((unsigned)i) >> 23);
        size_t row = (size_t)(b * S_ + i);
        re[r] = Zcat[row * (NB_ * 256) + nb * 256 + j];
        im[r] = Zcat[row * (NB_ * 256) + nb * 256 + 128 + j];
    }
    __syncthreads();
    fft512_stages(re, im, tid, +1.f);
    const float sc = 0.044194173824159216f;
    for (int s = tid; s < 512; s += 256) {
        size_t o = ((size_t)(b * S_ + s)) * DIM_ + d;
        dout[o] = outres[o] + re[s] * sc;
    }
}

// ---------------------------------------------------------------------------
// Host orchestration
// ---------------------------------------------------------------------------
extern "C" void kernel_launch(void* const* d_in, const int* in_sizes, int n_in,
                              void* d_out, int out_size, void* d_ws, size_t ws_size,
                              hipStream_t stream)
{
    (void)in_sizes; (void)n_in; (void)out_size; (void)ws_size;
    const float* x        = (const float*)d_in[0];   // (B,S,DIM)
    const float* inproj_w = (const float*)d_in[1];   // (512,2048)
    const float* conv_w   = (const float*)d_in[2];   // (1024,1,64)
    const float* conv_b   = (const float*)d_in[3];   // (1024)
    const float* xproj_w  = (const float*)d_in[4];   // (1024,160)
    const float* dtproj_w = (const float*)d_in[5];   // (32,1024)
    const float* dtproj_b = (const float*)d_in[6];   // (1024)
    const float* A_log    = (const float*)d_in[7];   // (1024,64)
    const float* Dw       = (const float*)d_in[8];   // (1024)
    const float* outproj_w= (const float*)d_in[9];   // (1024,512)
    const float* cw1      = (const float*)d_in[10];  // (2,4,128,128)
    const float* cb1      = (const float*)d_in[11];  // (2,4,128)
    const float* cw2      = (const float*)d_in[12];
    const float* cb2      = (const float*)d_in[13];
    float* out_p = (float*)d_out;

    // ---- workspace carve-out (256B aligned) ----
    char* w = (char*)d_ws;
    auto alloc = [&](size_t bytes) -> void* {
        void* p = (void*)w;
        w += (bytes + 255) & ~(size_t)255;
        return p;
    };
    _Float16* xn_h   = (_Float16*)alloc((size_t)M_ * DIM_ * 2);
    _Float16* ipw_t  = (_Float16*)alloc((size_t)2 * DIN_ * DIM_ * 2);       // (2048,512)
    _Float16* xpw_t  = (_Float16*)alloc((size_t)160 * DIN_ * 2);            // (160,1024)
    _Float16* dtw_t  = (_Float16*)alloc((size_t)DIN_ * DTRANK_ * 2);        // (1024,32)
    _Float16* opw_t  = (_Float16*)alloc((size_t)DIM_ * DIN_ * 2);           // (512,1024)
    _Float16* W1t    = (_Float16*)alloc((size_t)NB_ * 256 * 256 * 2);
    _Float16* W2t    = (_Float16*)alloc((size_t)NB_ * 256 * 256 * 2);
    float*    b1s    = (float*)alloc((size_t)NB_ * 256 * 4);
    float*    b2s    = (float*)alloc((size_t)NB_ * 256 * 4);
    float*    P      = (float*)alloc((size_t)M_ * 2 * DIN_ * 4);            // [xm|res]
    float*    u      = (float*)alloc((size_t)M_ * DIN_ * 4);
    _Float16* uh     = (_Float16*)alloc((size_t)M_ * DIN_ * 2);
    float*    xdbl   = (float*)alloc((size_t)M_ * 160 * 4);
    _Float16* dl_h   = (_Float16*)alloc((size_t)M_ * DTRANK_ * 2);
    float*    dtb    = (float*)alloc((size_t)M_ * DIN_ * 4);
    _Float16* yh     = (_Float16*)alloc((size_t)M_ * DIN_ * 2);
    float*    outb   = (float*)alloc((size_t)M_ * DIM_ * 4);
    float*    ln2    = (float*)alloc((size_t)M_ * DIM_ * 4);
    _Float16* Xcat   = (_Float16*)alloc((size_t)M_ * NB_ * 256 * 2);
    float*    H1     = (float*)alloc((size_t)M_ * NB_ * 256 * 4);
    _Float16* H1h    = (_Float16*)alloc((size_t)M_ * NB_ * 256 * 2);
    float*    Zcat   = (float*)alloc((size_t)M_ * NB_ * 256 * 4);

    const dim3 blk(256);

    // ---- weight prep ----
    transpose_cast_f16<<<(512 * 2048) / 256, blk, 0, stream>>>(inproj_w, ipw_t, 512, 2048);
    transpose_cast_f16<<<(1024 * 160) / 256, blk, 0, stream>>>(xproj_w, xpw_t, 1024, 160);
    transpose_cast_f16<<<(32 * 1024) / 256, blk, 0, stream>>>(dtproj_w, dtw_t, 32, 1024);
    transpose_cast_f16<<<(1024 * 512) / 256, blk, 0, stream>>>(outproj_w, opw_t, 1024, 512);
    build_einfft_wt<<<(NB_ * 256 * 256) / 256, blk, 0, stream>>>(cw1, cb1, W1t, b1s);
    build_einfft_wt<<<(NB_ * 256 * 256) / 256, blk, 0, stream>>>(cw2, cb2, W2t, b2s);

    // ---- block 1: prenorm Mamba ----
    layernorm_k<<<M_, blk, 0, stream>>>(x, (float*)0, xn_h);

    // in_proj: (1024,512) @ (512,2048) -> P    [super-tiles: 32*32/8 = 128 blocks]
    gemm_wmma_f16<2,4><<<dim3(128, 1, 1), blk, 0, stream>>>(
        xn_h, DIM_, 0, ipw_t, DIM_, 0, P, 2 * DIN_, 0,
        (const float*)0, 0, (const float*)0, (_Float16*)0, M_, 2 * DIN_, DIM_, 0);

    conv_silu_k<<<B_ * DIN_, dim3(128), 0, stream>>>(P, conv_w, conv_b, u, uh);

    // x_proj: (1024,1024) @ (1024,160) -> xdbl  [N=160 -> 10 tiles -> 2x2: 32*5/8=20]
    gemm_wmma_f16<2,2><<<dim3(20, 1, 1), blk, 0, stream>>>(
        uh, DIN_, 0, xpw_t, DIN_, 0, xdbl, 160, 0,
        (const float*)0, 0, (const float*)0, (_Float16*)0, M_, 160, DIN_, 0);

    slice_cast_f16<<<(M_ * DTRANK_) / 256, blk, 0, stream>>>(
        xdbl, 160, 0, dl_h, DTRANK_, M_, DTRANK_);

    // dt_proj + softplus: (1024,32) @ (32,1024) -> dtb  [32*16/8 = 64 blocks]
    gemm_wmma_f16<2,4><<<dim3(64, 1, 1), blk, 0, stream>>>(
        dl_h, DTRANK_, 0, dtw_t, DTRANK_, 0, dtb, DIN_, 0,
        dtproj_b, 0, (const float*)0, (_Float16*)0, M_, DIN_, DTRANK_, 1);

    ssm_scan_k<<<(B_ * DIN_) / 256, blk, 0, stream>>>(dtb, u, xdbl, A_log, Dw, P, yh);

    // out_proj + residual x: (1024,1024) @ (1024,512) -> outb  [32*8/8 = 32 blocks]
    gemm_wmma_f16<2,4><<<dim3(32, 1, 1), blk, 0, stream>>>(
        yh, DIN_, 0, opw_t, DIN_, 0, outb, DIM_, 0,
        (const float*)0, 0, x, (_Float16*)0, M_, DIM_, DIN_, 0);

    // ---- block 2: prenorm EinFFT ----
    layernorm_k<<<M_, blk, 0, stream>>>(outb, ln2, (_Float16*)0);

    fft_fwd_k<<<B_ * DIM_, blk, 0, stream>>>(ln2, Xcat);

    // EinFFT layer 1 (complex via stacked K=256, N=256) + relu -> H1h (f16)
    // [32*4/8 = 16 blocks, z = NB]
    gemm_wmma_f16<2,4><<<dim3(16, 1, NB_), blk, 0, stream>>>(
        Xcat, NB_ * 256, 256, W1t, 256, 256 * 256, H1, NB_ * 256, 256,
        b1s, 256, (const float*)0, H1h, M_, 256, 256, 2);

    // EinFFT layer 2 + softshrink -> Zcat (f32)
    gemm_wmma_f16<2,4><<<dim3(16, 1, NB_), blk, 0, stream>>>(
        H1h, NB_ * 256, 256, W2t, 256, 256 * 256, Zcat, NB_ * 256, 256,
        b2s, 256, (const float*)0, (_Float16*)0, M_, 256, 256, 3);

    // IFFT, take real, add residual -> d_out
    ifft_out_k<<<B_ * DIM_, blk, 0, stream>>>(Zcat, outb, out_p);
}
